// BiLSTMModel_80814104641854
// MI455X (gfx1250) — matrix operand order repaired and checked by reading, not compiled, not run on hardware
//
#include <hip/hip_runtime.h>
#include <hip/hip_bf16.h>
#include <stdint.h>

// ---------------------------------------------------------------------------
// BiLSTM (L=2, bidirectional) + FC for MI455X / gfx1250.
//   B=512, T=1024, IN=128, H=64, G=4H=256.
// Strategy:
//   * prep kernel: f32->bf16 weight conversion, pre-swizzled into WMMA
//     B-fragment lane layout (32B contiguous per lane per fragment); bias sum.
//   * lstm kernel: one workgroup (8 waves) per (16-row batch tile, direction).
//     Persistent t-loop; per step fuses input projection (K=128) and
//     recurrence (K=64) as v_wmma_f32_16x16x32_bf16, gates exchanged via LDS,
//     elementwise LSTM update, bf16 h written back for the next step.
//     Layer-1 input tiles staged via global_load_async_to_lds_b128.
//   * fc kernel: 128-wide dot per batch row.
// ---------------------------------------------------------------------------

#define B_  512
#define T_  1024
#define IN_ 128
#define H_  64
#define G_  256

typedef __attribute__((ext_vector_type(16))) __bf16 v16bf;
typedef __attribute__((ext_vector_type(8)))  float  v8f;
typedef __attribute__((ext_vector_type(16))) unsigned short us16;
typedef __attribute__((ext_vector_type(8)))  unsigned short us8;

// ---- workspace layout (bytes); requires ws_size >= ~135 MB -----------------
// w_ih swizzled: 4 layer-dirs * 16 ntiles * 4 ktiles * 512 halfs
#define WIH_OFF   0
#define WIH_HALFS (4 * 16 * 4 * 512)            // 131072
// w_hh swizzled: 4 layer-dirs * 16 ntiles * 2 ktiles * 512 halfs
#define WHH_OFF   (WIH_OFF + WIH_HALFS * 2)     // 262144
#define WHH_HALFS (4 * 16 * 2 * 512)            // 65536
#define BIAS_OFF  (WHH_OFF + WHH_HALFS * 2)     // 393216   (1024 f32)
#define LASTH_OFF (BIAS_OFF + 1024 * 4)         // 397312   (512*128 f32)
#define H0_OFF    (LASTH_OFF + B_ * 128 * 4 + 512) // bf16 (B,T,128)

__device__ __forceinline__ unsigned short f2bf(float f) {
  unsigned u = __builtin_bit_cast(unsigned, f);
  u += 0x7FFFu + ((u >> 16) & 1u);              // round-to-nearest-even
  return (unsigned short)(u >> 16);
}

__device__ __forceinline__ float fsigmoid(float x) {
  return 1.0f / (1.0f + __expf(-x));
}
__device__ __forceinline__ float ftanh(float x) {
  return 2.0f * fsigmoid(2.0f * x) - 1.0f;      // cheap tanh via v_exp_f32
}

// ---- CDNA5 async global->LDS copy (CDNA5 ISA) --------------
__device__ __forceinline__ void async_copy_b128(unsigned lds_off, const void* g) {
  asm volatile("global_load_async_to_lds_b128 %0, %1, off"
               :: "v"(lds_off), "v"((unsigned long long)(uintptr_t)g)
               : "memory");
}
__device__ __forceinline__ void wait_asynccnt0() {
  asm volatile("s_wait_asynccnt 0" ::: "memory");
}

// ---- fragment loaders ------------------------------------------------------
// A fragment, 16x32 bf16, from row-major LDS tile (stride halfs).
// ISA layout: lanes 0-15 row m=lane, hw0-7 -> k=0..7, hw8-15 -> k=16..23;
// lanes 16-31 same rows, k offset +8 / +24.  Two contiguous 16B chunks.
__device__ __forceinline__ v16bf load_afrag(const unsigned short* tile,
                                            int stride, int m, int kt,
                                            int lane_lt16) {
  int k0 = kt * 32 + (lane_lt16 ? 0 : 8);
  us8 lo = *(const us8*)(tile + m * stride + k0);
  us8 hi = *(const us8*)(tile + m * stride + k0 + 16);
  us16 a = __builtin_shufflevector(lo, hi, 0,1,2,3,4,5,6,7,8,9,10,11,12,13,14,15);
  return __builtin_bit_cast(v16bf, a);
}
// B fragment, pre-swizzled: lane reads 32 contiguous bytes.
__device__ __forceinline__ v16bf load_bfrag(const unsigned short* p) {
  return __builtin_bit_cast(v16bf, *(const us16*)p);
}

// ---------------------------------------------------------------------------
// prep: convert & swizzle weights, sum biases.
// ---------------------------------------------------------------------------
__global__ __launch_bounds__(256)
void prep_kernel(const float* __restrict__ w_ih, const float* __restrict__ w_hh,
                 const float* __restrict__ b_ih, const float* __restrict__ b_hh,
                 unsigned short* __restrict__ wih_s,
                 unsigned short* __restrict__ whh_s,
                 float* __restrict__ bias) {
  int idx = blockIdx.x * 256 + threadIdx.x;
  if (idx < WIH_HALFS) {                               // w_ih: K=128, 4 ktiles
    int ld = idx / 32768, r = idx % 32768;
    int nt = r / 2048;  r %= 2048;
    int kt = r / 512;   int h = r % 512;
    int lane = h / 16, hw = h % 16;
    int n = nt * 16 + (lane & 15);
    int k = kt * 32 + ((lane < 16) ? 0 : 8) + ((hw < 8) ? hw : 8 + hw);
    wih_s[idx] = f2bf(w_ih[((size_t)ld * G_ + n) * IN_ + k]);
    return;
  }
  idx -= WIH_HALFS;
  if (idx < WHH_HALFS) {                               // w_hh: K=64, 2 ktiles
    int ld = idx / 16384, r = idx % 16384;
    int nt = r / 1024;  r %= 1024;
    int kt = r / 512;   int h = r % 512;
    int lane = h / 16, hw = h % 16;
    int n = nt * 16 + (lane & 15);
    int k = kt * 32 + ((lane < 16) ? 0 : 8) + ((hw < 8) ? hw : 8 + hw);
    whh_s[idx] = f2bf(w_hh[((size_t)ld * G_ + n) * H_ + k]);
    return;
  }
  idx -= WHH_HALFS;
  if (idx < 4 * G_) bias[idx] = b_ih[idx] + b_hh[idx];
}

// ---------------------------------------------------------------------------
// One bidirectional LSTM layer.  grid = (B/16, 2 dirs), block = 256 (8 waves).
// ---------------------------------------------------------------------------
__global__ __launch_bounds__(256)
void lstm_layer_kernel(const float* __restrict__ xin,          // layer 0 input
                       const unsigned short* __restrict__ hin, // layer 1 input (bf16)
                       const unsigned short* __restrict__ wih_s,
                       const unsigned short* __restrict__ whh_s,
                       const float* __restrict__ bias,
                       unsigned short* __restrict__ hout,      // layer 0 output (bf16)
                       float* __restrict__ lasth,              // layer 1: t=T-1 slice
                       int layer) {
  __shared__ unsigned short xtile[16 * 128];   // input tile, bf16 row-major
  __shared__ unsigned short htile[16 * 64];    // hidden tile, bf16 row-major
  __shared__ float          gates[G_ * 16];    // [n][m] accumulators

  const int tid  = threadIdx.x;
  const int lane = tid & 31;
  const int wid  = tid >> 5;
  const int dir  = blockIdx.y;
  const int b0   = blockIdx.x * 16;
  const int ld   = layer * 2 + dir;

  const int mrow = lane & 15;
  const int llt  = (lane < 16);
  const int m0   = llt ? 0 : 8;

  // per-thread cell state: cells tid*4 .. tid*4+3 of the 16x64 (m,j) grid
  float creg[4] = {0.f, 0.f, 0.f, 0.f};
  // zero h
  #pragma unroll
  for (int q = 0; q < 4; ++q) htile[tid * 4 + q] = 0;
  __syncthreads();

  // staging geometry: thread copies 8 halfs (one 16B chunk) of the x tile
  const int srow = tid / 16;
  const int sk0  = (tid % 16) * 8;

  const float bval0 = bias[ld * G_ + (wid * 2 + 0) * 16 + (lane & 15)];
  const float bval1 = bias[ld * G_ + (wid * 2 + 1) * 16 + (lane & 15)];

  for (int s = 0; s < T_; ++s) {
    const int t = dir ? (T_ - 1 - s) : s;

    // ---- stage input tile (16 x 128) for time t --------------------------
    if (layer == 0) {
      const float* src = xin + (((size_t)(b0 + srow)) * T_ + t) * IN_ + sk0;
      float4 a = *(const float4*)src;
      float4 b = *(const float4*)(src + 4);
      us8 v = { f2bf(a.x), f2bf(a.y), f2bf(a.z), f2bf(a.w),
                f2bf(b.x), f2bf(b.y), f2bf(b.z), f2bf(b.w) };
      *(us8*)(&xtile[srow * 128 + sk0]) = v;
    } else {
      const unsigned short* src =
          hin + (((size_t)(b0 + srow)) * T_ + t) * 128 + sk0;
      unsigned lds_off = (unsigned)(uintptr_t)(&xtile[srow * 128 + sk0]);
      async_copy_b128(lds_off, src);
      wait_asynccnt0();
    }
    __syncthreads();

    // ---- gates = bias + x_t @ w_ih^T + h @ w_hh^T  (M16 N256 K128+64) ----
    #pragma unroll
    for (int i = 0; i < 2; ++i) {
      const int nt = wid * 2 + i;
      const float bv = i ? bval1 : bval0;
      v8f acc = {bv, bv, bv, bv, bv, bv, bv, bv};

      const unsigned short* wf =
          wih_s + ((size_t)(ld * 16 + nt) * 4) * 512 + lane * 16;
      #pragma unroll
      for (int kt = 0; kt < 4; ++kt) {
        v16bf a = load_afrag(xtile, 128, mrow, kt, llt);
        v16bf b = load_bfrag(wf + kt * 512);
        acc = __builtin_amdgcn_wmma_f32_16x16x32_bf16(
                  false, a, false, b, (short)0, acc, false, false);
      }
      const unsigned short* hf =
          whh_s + ((size_t)(ld * 16 + nt) * 2) * 512 + lane * 16;
      #pragma unroll
      for (int kt = 0; kt < 2; ++kt) {
        v16bf a = load_afrag(htile, 64, mrow, kt, llt);
        v16bf b = load_bfrag(hf + kt * 512);
        acc = __builtin_amdgcn_wmma_f32_16x16x32_bf16(
                  false, a, false, b, (short)0, acc, false, false);
      }
      // C layout: vgpr r -> m = m0 + r, n = nt*16 + (lane&15); store [n][m]
      const int n = nt * 16 + (lane & 15);
      float4 lo = make_float4(acc[0], acc[1], acc[2], acc[3]);
      float4 hi = make_float4(acc[4], acc[5], acc[6], acc[7]);
      *(float4*)(&gates[n * 16 + m0])     = lo;
      *(float4*)(&gates[n * 16 + m0 + 4]) = hi;
    }
    __syncthreads();

    // ---- elementwise LSTM update ----------------------------------------
    #pragma unroll
    for (int q = 0; q < 4; ++q) {
      const int cell = tid * 4 + q;
      const int m = cell >> 6;
      const int j = cell & 63;
      float iv = fsigmoid(gates[(      j) * 16 + m]);
      float fv = fsigmoid(gates[( 64 + j) * 16 + m]);
      float gv = ftanh   (gates[(128 + j) * 16 + m]);
      float ov = fsigmoid(gates[(192 + j) * 16 + m]);
      float c  = fv * creg[q] + iv * gv;
      creg[q]  = c;
      float h  = ov * ftanh(c);
      htile[m * 64 + j] = f2bf(h);
      if (layer == 0) {
        hout[(((size_t)(b0 + m)) * T_ + t) * 128 + dir * 64 + j] = f2bf(h);
      } else if (t == T_ - 1) {
        lasth[(size_t)(b0 + m) * 128 + dir * 64 + j] = h;
      }
    }
    __syncthreads();
  }
}

// ---------------------------------------------------------------------------
__global__ __launch_bounds__(256)
void fc_kernel(const float* __restrict__ lasth, const float* __restrict__ fcw,
               const float* __restrict__ fcb, float* __restrict__ out) {
  int b = blockIdx.x * 256 + threadIdx.x;
  if (b >= B_) return;
  float s = fcb[0];
  #pragma unroll 8
  for (int k = 0; k < 128; ++k) s += lasth[(size_t)b * 128 + k] * fcw[k];
  out[b] = s;
}

// ---------------------------------------------------------------------------
extern "C" void kernel_launch(void* const* d_in, const int* in_sizes, int n_in,
                              void* d_out, int out_size, void* d_ws, size_t ws_size,
                              hipStream_t stream) {
  const float* x    = (const float*)d_in[0];
  const float* w_ih = (const float*)d_in[1];
  const float* w_hh = (const float*)d_in[2];
  const float* b_ih = (const float*)d_in[3];
  const float* b_hh = (const float*)d_in[4];
  const float* fc_w = (const float*)d_in[5];
  const float* fc_b = (const float*)d_in[6];

  char* ws = (char*)d_ws;
  unsigned short* wih_s = (unsigned short*)(ws + WIH_OFF);
  unsigned short* whh_s = (unsigned short*)(ws + WHH_OFF);
  float*          bias  = (float*)(ws + BIAS_OFF);
  float*          lasth = (float*)(ws + LASTH_OFF);
  unsigned short* h0    = (unsigned short*)(ws + H0_OFF);

  const int prep_total = WIH_HALFS + WHH_HALFS + 4 * G_;
  prep_kernel<<<(prep_total + 255) / 256, 256, 0, stream>>>(
      w_ih, w_hh, b_ih, b_hh, wih_s, whh_s, bias);

  dim3 grid(B_ / 16, 2);
  lstm_layer_kernel<<<grid, 256, 0, stream>>>(
      x, nullptr, wih_s, whh_s, bias, h0, nullptr, 0);
  lstm_layer_kernel<<<grid, 256, 0, stream>>>(
      nullptr, h0, wih_s, whh_s, bias, nullptr, lasth, 1);

  fc_kernel<<<(B_ + 255) / 256, 256, 0, stream>>>(lasth, fc_w, fc_b, (float*)d_out);
}
